// SqueezeformerBlock_76493367542389
// MI455X (gfx1250) — matrix operand order repaired and checked by reading, not compiled, not run on hardware
//
#include <hip/hip_runtime.h>

#define NB 16
#define TT 512
#define DD 512
#define NH 8
#define DHD 64
#define FF 2048
#define KW 31
#define NT (NB*TT)          // 8192 rows
#define EPSF 1e-5f

typedef unsigned short u16;
typedef __attribute__((ext_vector_type(16))) __bf16 v16bf;
typedef __attribute__((ext_vector_type(8)))  __bf16 v8bf;
typedef __attribute__((ext_vector_type(8)))  float  v8f;
typedef __attribute__((ext_vector_type(8)))  unsigned short us8;
typedef __attribute__((ext_vector_type(4)))  unsigned int u32x4;
typedef __attribute__((ext_vector_type(8)))  unsigned int u32x8;

// ---------- helpers ----------
static __device__ __forceinline__ u16 f2bf(float f) {
  unsigned int u = __float_as_uint(f);
  u += 0x7fffu + ((u >> 16) & 1u);          // round-to-nearest-even
  return (u16)(u >> 16);
}

static __device__ __forceinline__ v16bf cat8(v8bf a, v8bf b) {
  return __builtin_shufflevector(a, b, 0,1,2,3,4,5,6,7,8,9,10,11,12,13,14,15);
}
// A-operand fragment: 16-bit A 16x32 layout, lane row = lane&15,
// K chunks {lh*8..lh*8+7, 16+lh*8..} (ISA 7.12.2)
static __device__ __forceinline__ v16bf frag_ilv(const u16* rowp, int lh) {
  v8bf a = *reinterpret_cast<const v8bf*>(rowp + lh * 8);
  v8bf b = *reinterpret_cast<const v8bf*>(rowp + 16 + lh * 8);
  return cat8(a, b);
}
// B-operand fragment: lane col = lane&15, K chunks {lh*8..lh*8+7, 16+lh*8..}
static __device__ __forceinline__ v16bf frag_ctg(const u16* rowp, int lh) {
  v8bf a = *reinterpret_cast<const v8bf*>(rowp + lh * 8);
  v8bf b = *reinterpret_cast<const v8bf*>(rowp + 16 + lh * 8);
  return cat8(a, b);
}
static __device__ __forceinline__ v8f wmma_bf16(v16bf a, v16bf b, v8f c) {
  return __builtin_amdgcn_wmma_f32_16x16x32_bf16(false, a, false, b, (short)0, c,
                                                 false, false);
}

// CDNA5 async global->LDS copy (GLOBAL_LOAD_ASYNC_TO_LDS_B128, ASYNCcnt).
// LDS generic pointers carry the LDS byte offset in their low 32 bits.
static __device__ __forceinline__ void async_ld_b128(void* lds_dst, const void* gsrc) {
  unsigned lds_off = (unsigned)(unsigned long long)lds_dst;
  asm volatile("global_load_async_to_lds_b128 %0, %1, off"
               :: "v"(lds_off), "v"(gsrc)
               : "memory");
}
static __device__ __forceinline__ void wait_async0() {
  asm volatile("s_wait_asynccnt 0x0" ::: "memory");
}

// CDNA5 Tensor Data Mover: DMA one 128-row x 32-elem bf16 tile (row stride
// `kstride` elems) from global into LDS, padding each 64B LDS row by 16B
// (pad_interval=3 -> every 16 DWORDs, pad_amount=3 -> 4 DWORDs) to reproduce
// the stride-40 LDS layout the WMMA fragments read. D# per ISA ch.8.
static __device__ __forceinline__ void tdm_load_tile(void* lds_dst,
                                                     const u16* gsrc,
                                                     int kstride) {
  unsigned long long ga = (unsigned long long)gsrc;
  u32x4 g0;
  g0[0] = 1u;                                           // count=1, no gather
  g0[1] = (unsigned)(unsigned long long)lds_dst;        // lds_addr (bytes)
  g0[2] = (unsigned)ga;                                 // global_addr[31:0]
  g0[3] = (unsigned)((ga >> 32) & 0x01ffffffull)        // global_addr[56:32]
        | (2u << 30);                                   // type=2 ("image")
  u32x8 g1;
  g1[0] = (1u << 16)        // data_size = 2 bytes
        | (1u << 20)        // pad_enable
        | (3u << 22)        // pad_interval: 16 DWORDs
        | (3u << 25);       // pad_amount: 4 DWORDs
  g1[1] = (32u  << 16);     // tensor_dim0[15:0]=32 (== tile: no OOB)
  g1[2] = (128u << 16);     // tensor_dim0 hi=0 | tensor_dim1[15:0]=128
  g1[3] = (32u  << 16);     // tensor_dim1 hi=0 | tile_dim0=32
  g1[4] = 128u;             // tile_dim1=128, tile_dim2=0
  g1[5] = (unsigned)kstride;// tensor_dim0_stride[31:0] (elems)
  g1[6] = 0u;
  g1[7] = 0u;
  asm volatile("tensor_load_to_lds %0, %1" :: "s"(g0), "s"(g1) : "memory");
}

// ---------- elementwise kernels ----------
__global__ __launch_bounds__(256) void k_f32_to_bf16(const float* __restrict__ in,
                                                     u16* __restrict__ out, int n) {
  int i = blockIdx.x * 256 + threadIdx.x;
  if (i < n) out[i] = f2bf(in[i]);
}

__global__ __launch_bounds__(256) void k_scale_bias(const float* __restrict__ x,
                                                    const float* __restrict__ s,
                                                    const float* __restrict__ b,
                                                    u16* __restrict__ out) {
  int i = blockIdx.x * 256 + threadIdx.x;
  int d = i & (DD - 1);
  out[i] = f2bf(x[i] * s[d] + b[d]);
}

__global__ __launch_bounds__(256) void k_swish_bf16(const float* __restrict__ in,
                                                    u16* __restrict__ out) {
  int i = blockIdx.x * 256 + threadIdx.x;
  float v = in[i];
  out[i] = f2bf(v / (1.f + __expf(-v)));
}

__global__ __launch_bounds__(256) void k_glu(const float* __restrict__ z,
                                             float* __restrict__ out) {
  int i = blockIdx.x * 256 + threadIdx.x;
  int row = i >> 9, d = i & 511;
  float a = z[(size_t)row * 1024 + d];
  float g = z[(size_t)row * 1024 + 512 + d];
  out[i] = a / (1.f + __expf(-g));
}

// qkv f32 [NT,1536] -> q/k/v bf16 [B,H,T,64] with RoPE on q,k
__global__ __launch_bounds__(256) void k_rope(const float* __restrict__ qkv,
                                              u16* __restrict__ q, u16* __restrict__ k,
                                              u16* __restrict__ v) {
  int i = blockIdx.x * 256 + threadIdx.x;   // (row, h, d2)
  int d2 = i & 31;
  int h = (i >> 5) & 7;
  int row = i >> 8;                          // b*T + t
  int t = row & (TT - 1);
  int b = row >> 9;
  float ang = (float)t * __powf(10000.f, -(float)d2 * (1.f / 32.f));
  float sn, cs;
  sincosf(ang, &sn, &cs);
  const float* base = qkv + (size_t)row * 1536 + h * 64 + d2;
  float q1 = base[0],    q2 = base[32];
  float k1 = base[512],  k2 = base[544];
  float v1 = base[1024], v2 = base[1056];
  size_t ob = ((size_t)(b * NH + h) * TT + t) * DHD + d2;
  q[ob]      = f2bf(q1 * cs - q2 * sn);
  q[ob + 32] = f2bf(q2 * cs + q1 * sn);
  k[ob]      = f2bf(k1 * cs - k2 * sn);
  k[ob + 32] = f2bf(k2 * cs + k1 * sn);
  v[ob]      = f2bf(v1);
  v[ob + 32] = f2bf(v2);
}

__global__ __launch_bounds__(256) void k_dwconv(const float* __restrict__ zg,
                                                const float* __restrict__ w,
                                                const float* __restrict__ bias,
                                                float* __restrict__ out) {
  int i = blockIdx.x * 256 + threadIdx.x;
  int d = i & 511, row = i >> 9;
  int t = row & (TT - 1), b = row >> 9;
  float acc = bias[d];
#pragma unroll
  for (int kk = 0; kk < KW; kk++) {
    int tt = t - 15 + kk;
    if (tt >= 0 && tt < TT) acc += zg[(size_t)(b * TT + tt) * DD + d] * w[d * KW + kk];
  }
  out[i] = acc;
}

__global__ __launch_bounds__(256) void k_bn_stats(const float* __restrict__ zc,
                                                  float* __restrict__ stats) {
  int d = blockIdx.x, tid = threadIdx.x;
  float s = 0.f, q = 0.f;
  for (int r = tid; r < NT; r += 256) {
    float v = zc[(size_t)r * DD + d];
    s += v; q += v * v;
  }
  __shared__ float rs[256], rq[256];
  rs[tid] = s; rq[tid] = q;
  __syncthreads();
  for (int o = 128; o > 0; o >>= 1) {
    if (tid < o) { rs[tid] += rs[tid + o]; rq[tid] += rq[tid + o]; }
    __syncthreads();
  }
  if (tid == 0) {
    float mean = rs[0] * (1.f / NT);
    stats[d] = mean;
    stats[DD + d] = rq[0] * (1.f / NT) - mean * mean;
  }
}

__global__ __launch_bounds__(256) void k_bn_swish(const float* __restrict__ zc,
                                                  const float* __restrict__ stats,
                                                  const float* __restrict__ g,
                                                  const float* __restrict__ b,
                                                  u16* __restrict__ out) {
  int i = blockIdx.x * 256 + threadIdx.x;
  int d = i & 511;
  float xn = (zc[i] - stats[d]) * rsqrtf(stats[DD + d] + EPSF) * g[d] + b[d];
  out[i] = f2bf(xn / (1.f + __expf(-xn)));
}

// out = LN(x + y); safe in-place (out may alias x): row fully read before write
__global__ __launch_bounds__(256) void k_residual_ln(const float* __restrict__ x,
                                                     const float* __restrict__ y,
                                                     const float* __restrict__ g,
                                                     const float* __restrict__ bb,
                                                     float* __restrict__ out) {
  __shared__ float red[256];
  int row = blockIdx.x, tid = threadIdx.x;
  const float* xr = x + (size_t)row * DD;
  const float* yr = y + (size_t)row * DD;
  float v0 = xr[tid] + yr[tid];
  float v1 = xr[tid + 256] + yr[tid + 256];
  red[tid] = v0 + v1;
  __syncthreads();
  for (int o = 128; o > 0; o >>= 1) { if (tid < o) red[tid] += red[tid + o]; __syncthreads(); }
  float mean = red[0] * (1.f / DD);
  __syncthreads();
  float d0 = v0 - mean, d1 = v1 - mean;
  red[tid] = d0 * d0 + d1 * d1;
  __syncthreads();
  for (int o = 128; o > 0; o >>= 1) { if (tid < o) red[tid] += red[tid + o]; __syncthreads(); }
  float inv = rsqrtf(red[0] * (1.f / DD) + EPSF);
  out[(size_t)row * DD + tid]       = d0 * inv * g[tid] + bb[tid];
  out[(size_t)row * DD + tid + 256] = d1 * inv * g[tid + 256] + bb[tid + 256];
}

// ---------- bf16 GEMM: C[M,N] = A[M,K]*B[K,N] (+bias) ----------
// 128x128 macro-tile, 8 waves (2x4), double-buffered LDS.
// A tile: one TDM tensor_load_to_lds per k-step (wave 0 issues, TENSORcnt).
// B tile: register-staged + transposed into LDS.
__global__ __launch_bounds__(256) void k_gemm_bf16(const u16* __restrict__ A,
                                                   const u16* __restrict__ Bw,
                                                   const float* __restrict__ bias,
                                                   float* __restrict__ C,
                                                   int M, int N, int Kd) {
  __shared__ u16 As[2][128 * 40];   // [row][k] stride 40 (TDM pad fills this)
  __shared__ u16 Bs[2][128 * 40];   // transposed: [n][k] stride 40
  const int tid = threadIdx.x;
  const int wid = tid >> 5, lane = tid & 31, lh = lane >> 4, r = lane & 15;
  const int n0 = blockIdx.x * 128, m0 = blockIdx.y * 128;
  const int wm = wid >> 2, wn = wid & 3;    // 2 x 4 wave grid -> 64x32 per wave
  v8f acc[4][2];
#pragma unroll
  for (int i = 0; i < 4; i++)
#pragma unroll
    for (int j = 0; j < 2; j++)
#pragma unroll
      for (int e = 0; e < 8; e++) acc[i][j][e] = 0.f;
  const int bk = tid >> 3, bn8 = (tid & 7) * 16;

  // prologue: kick TDM for tile 0 (one DMA covers the whole 128x32 A tile)
  if (wid == 0) tdm_load_tile(&As[0][0], A + (size_t)m0 * Kd, Kd);
  const u16* bg = Bw + (size_t)bk * N + n0 + bn8;
  us8 breg0 = *reinterpret_cast<const us8*>(bg);
  us8 breg1 = *reinterpret_cast<const us8*>(bg + 8);

  const int nsteps = Kd >> 5;
  for (int s = 0; s < nsteps; s++) {
    const int cur = s & 1;
    // commit current tile: issuing wave drains TENSORcnt; everyone scatters B
    __builtin_amdgcn_s_wait_tensorcnt(0);
    u16* bs = &Bs[cur][0];
#pragma unroll
    for (int e = 0; e < 8; e++) {
      bs[(bn8 + e) * 40 + bk]     = breg0[e];
      bs[(bn8 + 8 + e) * 40 + bk] = breg1[e];
    }
    __syncthreads();                       // tile `cur` ready for all waves
    // prefetch next tile into the other buffer while we compute
    if (s + 1 < nsteps) {
      const int kt = (s + 1) << 5;
      if (wid == 0)
        tdm_load_tile(&As[cur ^ 1][0], A + (size_t)m0 * Kd + kt, Kd);
      const u16* bgn = Bw + (size_t)(kt + bk) * N + n0 + bn8;
      breg0 = *reinterpret_cast<const us8*>(bgn);
      breg1 = *reinterpret_cast<const us8*>(bgn + 8);
    }
    v16bf afr[4], bfr[2];
#pragma unroll
    for (int i = 0; i < 4; i++)
      afr[i] = frag_ilv(&As[cur][(wm * 64 + i * 16 + r) * 40], lh);
#pragma unroll
    for (int j = 0; j < 2; j++)
      bfr[j] = frag_ctg(&Bs[cur][(wn * 32 + j * 16 + r) * 40], lh);
#pragma unroll
    for (int i = 0; i < 4; i++)
#pragma unroll
      for (int j = 0; j < 2; j++) acc[i][j] = wmma_bf16(afr[i], bfr[j], acc[i][j]);
    __syncthreads();                       // reads of `cur` done before reuse
  }
#pragma unroll
  for (int i = 0; i < 4; i++)
#pragma unroll
    for (int j = 0; j < 2; j++) {
      int col = n0 + wn * 32 + j * 16 + r;
      float bv = bias ? bias[col] : 0.f;
#pragma unroll
      for (int e = 0; e < 8; e++) {
        int rowg = m0 + wm * 64 + i * 16 + e + 8 * lh;   // C layout: e -> M=e+8*half
        C[(size_t)rowg * N + col] = acc[i][j][e] + bv;
      }
    }
}

// ---------- flash attention: one block per (b,h,128 q rows), softmax(QK^T/8)V ----------
__global__ __launch_bounds__(256) void k_attn(const u16* __restrict__ Q,
                                              const u16* __restrict__ K,
                                              const u16* __restrict__ V,
                                              float* __restrict__ O) {
  __shared__ u16 Ks[32 * 72];        // [key][dh]   stride 72 (async-filled)
  __shared__ u16 Vs[64 * 40];        // [dh][key]   stride 40 (transposed)
  __shared__ u16 Ps[8 * 16 * 40];    // per-wave P  [qrow][key] stride 40
  const int tid = threadIdx.x;
  const int wid = tid >> 5, lane = tid & 31, lh = lane >> 4, r = lane & 15;
  const int qb = blockIdx.x, h = blockIdx.y, b = blockIdx.z;
  const int bh = b * NH + h;
  const u16* Qb = Q + (size_t)bh * TT * DHD;
  const u16* Kb = K + (size_t)bh * TT * DHD;
  const u16* Vb = V + (size_t)bh * TT * DHD;
  const int q0 = qb * 128 + wid * 16;
  const u16* qrow = Qb + (size_t)(q0 + r) * DHD;
  v16bf qf0 = frag_ilv(qrow, lh);          // dh 0..31
  v16bf qf1 = frag_ilv(qrow + 32, lh);     // dh 32..63
  v8f oacc[4];
  float m[8], l[8];
#pragma unroll
  for (int e = 0; e < 8; e++) {
    m[e] = -1e30f; l[e] = 0.f;
#pragma unroll
    for (int n = 0; n < 4; n++) oacc[n][e] = 0.f;
  }
  const int key = tid >> 3, ch = tid & 7;
  for (int k0 = 0; k0 < TT; k0 += 32) {
    // K tile: direct async copy; V tile: register transpose
    async_ld_b128(&Ks[key * 72 + ch * 8],
                  Kb + (size_t)(k0 + key) * DHD + ch * 8);
    us8 vd = *reinterpret_cast<const us8*>(Vb + (size_t)(k0 + key) * DHD + ch * 8);
#pragma unroll
    for (int e = 0; e < 8; e++) Vs[(ch * 8 + e) * 40 + key] = vd[e];
    wait_async0();
    __syncthreads();
    v8f s0, s1;
#pragma unroll
    for (int e = 0; e < 8; e++) { s0[e] = 0.f; s1[e] = 0.f; }
    const u16* kp0 = &Ks[r * 72];
    const u16* kp1 = &Ks[(16 + r) * 72];
    v16bf kb00 = frag_ctg(kp0, lh),      kb01 = frag_ctg(kp0 + 32, lh);
    v16bf kb10 = frag_ctg(kp1, lh),      kb11 = frag_ctg(kp1 + 32, lh);
    s0 = wmma_bf16(qf0, kb00, s0);
    s0 = wmma_bf16(qf1, kb01, s0);
    s1 = wmma_bf16(qf0, kb10, s1);
    s1 = wmma_bf16(qf1, kb11, s1);
    // online softmax across 32 keys; row stats per (e, half) via width-16 shuffles
#pragma unroll
    for (int e = 0; e < 8; e++) {
      float a = s0[e] * 0.125f, c = s1[e] * 0.125f;   // DH^-0.5
      float mx = fmaxf(a, c);
#pragma unroll
      for (int off = 1; off < 16; off <<= 1) mx = fmaxf(mx, __shfl_xor(mx, off, 16));
      float mn = fmaxf(m[e], mx);
      float resc = __expf(m[e] - mn);
      m[e] = mn;
      float ea = __expf(a - mn), ec = __expf(c - mn);
      float rs = ea + ec;
#pragma unroll
      for (int off = 1; off < 16; off <<= 1) rs += __shfl_xor(rs, off, 16);
      l[e] = l[e] * resc + rs;
#pragma unroll
      for (int n = 0; n < 4; n++) oacc[n][e] *= resc;
      int prow = e + 8 * lh;
      Ps[wid * 640 + prow * 40 + r]      = f2bf(ea);   // key col r
      Ps[wid * 640 + prow * 40 + 16 + r] = f2bf(ec);   // key col 16+r
    }
    // wave-private LDS round-trip converts C layout -> A layout
    v16bf pf = frag_ilv(&Ps[wid * 640 + r * 40], lh);
#pragma unroll
    for (int n = 0; n < 4; n++) {
      v16bf vf = frag_ctg(&Vs[(n * 16 + r) * 40], lh);
      oacc[n] = wmma_bf16(pf, vf, oacc[n]);
    }
    __syncthreads();
  }
#pragma unroll
  for (int e = 0; e < 8; e++) {
    float invl = 1.f / l[e];
    int qg = q0 + e + 8 * lh;
#pragma unroll
    for (int n = 0; n < 4; n++)
      O[(size_t)(b * TT + qg) * DD + h * DHD + n * 16 + r] = oacc[n][e] * invl;
  }
}

// ---------- host ----------
extern "C" void kernel_launch(void* const* d_in, const int* in_sizes, int n_in,
                              void* d_out, int out_size, void* d_ws, size_t ws_size,
                              hipStream_t stream) {
  (void)in_sizes; (void)n_in; (void)out_size; (void)ws_size;
  const float* x    = (const float*)d_in[0];
  const float* sb1s = (const float*)d_in[1];
  const float* sb1b = (const float*)d_in[2];
  const float* ln1g = (const float*)d_in[3];
  const float* ln1b = (const float*)d_in[4];
  const float* sb2s = (const float*)d_in[5];
  const float* sb2b = (const float*)d_in[6];
  const float* ln2g = (const float*)d_in[7];
  const float* ln2b = (const float*)d_in[8];
  const float* sb3s = (const float*)d_in[9];
  const float* sb3b = (const float*)d_in[10];
  const float* ln3g = (const float*)d_in[11];
  const float* ln3b = (const float*)d_in[12];
  const float* sb4s = (const float*)d_in[13];
  const float* sb4b = (const float*)d_in[14];
  const float* ln4g = (const float*)d_in[15];
  const float* ln4b = (const float*)d_in[16];
  const float* w_qkv = (const float*)d_in[17];
  const float* w_out = (const float*)d_in[18];
  const float* f1w1 = (const float*)d_in[19];
  const float* f1b1 = (const float*)d_in[20];
  const float* f1w2 = (const float*)d_in[21];
  const float* f1b2 = (const float*)d_in[22];
  const float* f2w1 = (const float*)d_in[23];
  const float* f2b1 = (const float*)d_in[24];
  const float* f2w2 = (const float*)d_in[25];
  const float* f2b2 = (const float*)d_in[26];
  const float* pw1w = (const float*)d_in[27];
  const float* pw1b = (const float*)d_in[28];
  const float* dww  = (const float*)d_in[29];
  const float* dwb  = (const float*)d_in[30];
  const float* bng  = (const float*)d_in[31];
  const float* bnb  = (const float*)d_in[32];
  const float* pw2w = (const float*)d_in[33];
  const float* pw2b = (const float*)d_in[34];
  float* out = (float*)d_out;

  char* ws = (char*)d_ws;
  size_t off = 0;
  auto take = [&](size_t bytes) -> char* {
    char* p = ws + off;
    off += (bytes + 255) & ~(size_t)255;
    return p;
  };
  u16* wb_qkv  = (u16*)take((size_t)DD * 3 * DD * 2);
  u16* wb_out  = (u16*)take((size_t)DD * DD * 2);
  u16* wb_f1w1 = (u16*)take((size_t)DD * FF * 2);
  u16* wb_f1w2 = (u16*)take((size_t)FF * DD * 2);
  u16* wb_pw1  = (u16*)take((size_t)DD * 2 * DD * 2);
  u16* wb_pw2  = (u16*)take((size_t)DD * DD * 2);
  u16* wb_f2w1 = (u16*)take((size_t)DD * FF * 2);
  u16* wb_f2w2 = (u16*)take((size_t)FF * DD * 2);
  float* xcur  = (float*)take((size_t)NT * DD * 4);
  float* bufY  = (float*)take((size_t)NT * DD * 4);
  float* bufH  = (float*)take((size_t)NT * FF * 4);
  u16*   bufA  = (u16*)take((size_t)NT * FF * 2);
  float* stats = (float*)take(2 * DD * 4);
  u16* q_bf = bufA + (size_t)NT * DD;       // bufA[0..NT*DD) reused per stage
  u16* k_bf = bufA + (size_t)NT * DD * 2;
  u16* v_bf = bufA + (size_t)NT * DD * 3;
  float* zg = bufH + (size_t)NT * (FF / 2); // GLU output [NT,DD]
  float* zc = zg + (size_t)NT * DD;         // dwconv output [NT,DD]

  dim3 blk(256);
  // weights -> bf16 (every call; deterministic)
  k_f32_to_bf16<<<(DD * 3 * DD) / 256, blk, 0, stream>>>(w_qkv, wb_qkv, DD * 3 * DD);
  k_f32_to_bf16<<<(DD * DD) / 256,     blk, 0, stream>>>(w_out, wb_out, DD * DD);
  k_f32_to_bf16<<<(DD * FF) / 256,     blk, 0, stream>>>(f1w1, wb_f1w1, DD * FF);
  k_f32_to_bf16<<<(FF * DD) / 256,     blk, 0, stream>>>(f1w2, wb_f1w2, FF * DD);
  k_f32_to_bf16<<<(DD * 2 * DD) / 256, blk, 0, stream>>>(pw1w, wb_pw1, DD * 2 * DD);
  k_f32_to_bf16<<<(DD * DD) / 256,     blk, 0, stream>>>(pw2w, wb_pw2, DD * DD);
  k_f32_to_bf16<<<(DD * FF) / 256,     blk, 0, stream>>>(f2w1, wb_f2w1, DD * FF);
  k_f32_to_bf16<<<(FF * DD) / 256,     blk, 0, stream>>>(f2w2, wb_f2w2, FF * DD);

  // ---- sub-block 1: MHSA (RoPE) ----
  k_scale_bias<<<(NT * DD) / 256, blk, 0, stream>>>(x, sb1s, sb1b, bufA);
  k_gemm_bf16<<<dim3(12, 64), blk, 0, stream>>>(bufA, wb_qkv, nullptr, bufH, NT, 3 * DD, DD);
  k_rope<<<(NT * NH * 32) / 256, blk, 0, stream>>>(bufH, q_bf, k_bf, v_bf);
  k_attn<<<dim3(TT / 128, NH, NB), blk, 0, stream>>>(q_bf, k_bf, v_bf, bufH);
  k_f32_to_bf16<<<(NT * DD) / 256, blk, 0, stream>>>(bufH, bufA, NT * DD);
  k_gemm_bf16<<<dim3(4, 64), blk, 0, stream>>>(bufA, wb_out, nullptr, bufY, NT, DD, DD);
  k_residual_ln<<<NT, blk, 0, stream>>>(x, bufY, ln1g, ln1b, xcur);

  // ---- sub-block 2: FFN1 ----
  k_scale_bias<<<(NT * DD) / 256, blk, 0, stream>>>(xcur, sb2s, sb2b, bufA);
  k_gemm_bf16<<<dim3(16, 64), blk, 0, stream>>>(bufA, wb_f1w1, f1b1, bufH, NT, FF, DD);
  k_swish_bf16<<<(NT * FF) / 256, blk, 0, stream>>>(bufH, bufA);
  k_gemm_bf16<<<dim3(4, 64), blk, 0, stream>>>(bufA, wb_f1w2, f1b2, bufY, NT, DD, FF);
  k_residual_ln<<<NT, blk, 0, stream>>>(xcur, bufY, ln2g, ln2b, xcur);

  // ---- sub-block 3: conv module ----
  k_scale_bias<<<(NT * DD) / 256, blk, 0, stream>>>(xcur, sb3s, sb3b, bufA);
  k_gemm_bf16<<<dim3(8, 64), blk, 0, stream>>>(bufA, wb_pw1, pw1b, bufH, NT, 2 * DD, DD);
  k_glu<<<(NT * DD) / 256, blk, 0, stream>>>(bufH, zg);
  k_dwconv<<<(NT * DD) / 256, blk, 0, stream>>>(zg, dww, dwb, zc);
  k_bn_stats<<<DD, blk, 0, stream>>>(zc, stats);
  k_bn_swish<<<(NT * DD) / 256, blk, 0, stream>>>(zc, stats, bng, bnb, bufA);
  k_gemm_bf16<<<dim3(4, 64), blk, 0, stream>>>(bufA, wb_pw2, pw2b, bufY, NT, DD, DD);
  k_residual_ln<<<NT, blk, 0, stream>>>(xcur, bufY, ln3g, ln3b, xcur);

  // ---- sub-block 4: FFN2 ----
  k_scale_bias<<<(NT * DD) / 256, blk, 0, stream>>>(xcur, sb4s, sb4b, bufA);
  k_gemm_bf16<<<dim3(16, 64), blk, 0, stream>>>(bufA, wb_f2w1, f2b1, bufH, NT, FF, DD);
  k_swish_bf16<<<(NT * FF) / 256, blk, 0, stream>>>(bufH, bufA);
  k_gemm_bf16<<<dim3(4, 64), blk, 0, stream>>>(bufA, wb_f2w2, f2b2, bufY, NT, DD, FF);
  k_residual_ln<<<NT, blk, 0, stream>>>(xcur, bufY, ln4g, ln4b, out);
}
